// ModulatedConv2d_64991445123922
// MI455X (gfx1250) — matrix operand order repaired and checked by reading, not compiled, not run on hardware
//
#include <hip/hip_runtime.h>
#include <hip/hip_bf16.h>

// ---------------------------------------------------------------------------
// StyleGAN2 ModulatedConv2d for MI455X (gfx1250, wave32, WMMA + TDM)
// B=16, CIN=COUT=512, K=3, H=W=64, SDIM=512
// Per-batch implicit GEMM: out[b](512x4096) = Wmod[b](512x4608) @ im2col(x[b])
// bf16 WMMA (16x16x32) with f32 accumulation, TDM-staged A tiles,
// double-buffered LDS pipeline.
// ---------------------------------------------------------------------------

typedef __attribute__((ext_vector_type(16))) __bf16        v16bf;
typedef __attribute__((ext_vector_type(8)))  float         v8f;
typedef __attribute__((ext_vector_type(4)))  unsigned int  u32x4;
typedef __attribute__((ext_vector_type(8)))  unsigned int  u32x8;

#define B_    16
#define CIN_  512
#define COUT_ 512
#define HW_   64
#define HP_   66          // padded rows   (h in [-1,64])
#define WP_   72          // padded cols   (w in [-1,64], padded to 72)
#define PLANE_ (HP_ * WP_)   // 4752 elements per padded channel plane
#define NCHUNK_ 144          // 9 taps * 16 cin-chunks of 32

__device__ __forceinline__ unsigned short f2bf(float f) {
    unsigned int u = __float_as_uint(f);
    unsigned int r = u + 0x7FFFu + ((u >> 16) & 1u);   // round-to-nearest-even
    return (unsigned short)(r >> 16);
}

// ---------------------------------------------------------------------------
// Kernel 1: s[b][cin] = style[b,:] . mod_w[cin,:] + mod_b[cin]
// ---------------------------------------------------------------------------
__global__ void style_mod_kernel(const float* __restrict__ style,
                                 const float* __restrict__ mod_w,
                                 const float* __restrict__ mod_b,
                                 float* __restrict__ s) {
    int idx = blockIdx.x * blockDim.x + threadIdx.x;   // 0..8191
    int b   = idx >> 9;
    int cin = idx & 511;
    const float* st = style + (long)b * 512;
    const float* mw = mod_w + (long)cin * 512;
    float acc = mod_b[cin];
    #pragma unroll 8
    for (int k = 0; k < 512; ++k) acc += st[k] * mw[k];
    s[idx] = acc;
}

// ---------------------------------------------------------------------------
// Kernel 2: demod[b][cout] = rsqrt( sum_{cin,kk} (w*s)^2 + 1e-8 ), wave per out
// ---------------------------------------------------------------------------
__global__ void demod_kernel(const float* __restrict__ weight,   // [COUT][CIN][9]
                             const float* __restrict__ s,        // [B][CIN]
                             float* __restrict__ demod) {        // [B][COUT]
    int wave = threadIdx.x >> 5;
    int lane = threadIdx.x & 31;
    int gid  = blockIdx.x * (blockDim.x >> 5) + wave;   // 0..8191
    int b    = gid >> 9;
    int cout = gid & 511;
    float acc = 0.0f;
    for (int cin = lane; cin < CIN_; cin += 32) {
        float sv = s[b * CIN_ + cin];
        const float* wp = weight + ((long)cout * CIN_ + cin) * 9;
        float ws = 0.0f;
        #pragma unroll
        for (int kk = 0; kk < 9; ++kk) { float wv = wp[kk]; ws += wv * wv; }
        acc += ws * sv * sv;
    }
    #pragma unroll
    for (int off = 16; off > 0; off >>= 1) acc += __shfl_xor(acc, off, 32);
    if (lane == 0) demod[gid] = rsqrtf(acc + 1e-8f);
}

// ---------------------------------------------------------------------------
// Kernel 3: wprep[b][tap][cout][cin] = bf16(w[cout][cin][tap] * s * demod)
// ---------------------------------------------------------------------------
__global__ void prep_weights_kernel(const float* __restrict__ weight,
                                    const float* __restrict__ s,
                                    const float* __restrict__ demod,
                                    unsigned short* __restrict__ wprep) {
    const long long total = (long long)B_ * 9 * COUT_ * CIN_;
    long long idx = (long long)blockIdx.x * blockDim.x + threadIdx.x;
    long long stride = (long long)gridDim.x * blockDim.x;
    for (; idx < total; idx += stride) {
        int cin  = (int)(idx & 511);
        int cout = (int)((idx >> 9) & 511);
        int tap  = (int)((idx / (512LL * 512LL)) % 9);
        int b    = (int)(idx / (512LL * 512LL * 9LL));
        float v = weight[((long)cout * CIN_ + cin) * 9 + tap]
                * s[b * CIN_ + cin] * demod[b * COUT_ + cout];
        wprep[idx] = f2bf(v);
    }
}

// ---------------------------------------------------------------------------
// Kernel 4: x -> zero-padded bf16  xpad[B][CIN][66][72]
// ---------------------------------------------------------------------------
__global__ void pad_x_kernel(const float* __restrict__ x,
                             unsigned short* __restrict__ xpad) {
    int plane = blockIdx.x;                       // b*CIN + c  (8192 planes)
    long obase = (long)plane * PLANE_;
    long ibase = (long)plane * (HW_ * HW_);
    for (int i = threadIdx.x; i < PLANE_; i += 256) {
        int r  = i / WP_;          // 0..65
        int cc = i - r * WP_;      // 0..71
        int h = r - 1, w = cc - 1;
        unsigned short v = 0;
        if (h >= 0 && h < HW_ && w >= 0 && w < HW_)
            v = f2bf(x[ibase + h * HW_ + w]);
        xpad[obase + i] = v;
    }
}

// ---------------------------------------------------------------------------
// Kernel 5: WMMA implicit GEMM, TDM A-staging, double-buffered LDS.
// Block = 256 threads (8 waves). Output tile: 128 couts x 128 pixels.
// Wave (wm, wn) owns 32 couts x 64 pixels = 2x4 accumulators.
// ---------------------------------------------------------------------------
__global__ __launch_bounds__(256)
void modconv_wmma_kernel(const unsigned short* __restrict__ xpad,  // [B][CIN][66][72] bf16
                         const unsigned short* __restrict__ wprep, // [B][9][COUT][CIN] bf16
                         const float* __restrict__ bias,
                         float* __restrict__ out) {                // [B][COUT][64][64]
    __shared__ __align__(16) unsigned short Alds[2][128 * 32];  // [buf][cout][cin]
    __shared__ __align__(16) unsigned short Blds[2][128 * 32];  // [buf][pixel][k]

    const int nt = blockIdx.x;          // 0..31 : pixel tile (2 image rows)
    const int mt = blockIdx.y;          // 0..3  : cout tile
    const int b  = blockIdx.z;          // 0..15
    const int mbase = mt * 128;
    const int h0    = nt * 2;

    const int tid  = threadIdx.x;
    const int lane = tid & 31;
    const int wave = tid >> 5;
    const int wm   = wave & 3;
    const int wn   = wave >> 2;
    const int lrow = lane & 15;
    const int lhi  = lane >> 4;

    v8f acc[2][4];
    #pragma unroll
    for (int mi = 0; mi < 2; ++mi)
        #pragma unroll
        for (int nj = 0; nj < 4; ++nj)
            #pragma unroll
            for (int v = 0; v < 8; ++v) acc[mi][nj][v] = 0.0f;

    // --- TDM: DMA the 128x32 bf16 A tile (global row-stride 512) into LDS ---
    auto tdmA = [&](int chunk, int buf) {
        int tap = chunk >> 4;
        int c0  = (chunk & 15) << 5;
        const unsigned short* src = wprep
            + (long)b * 9 * COUT_ * CIN_ + (long)tap * COUT_ * CIN_
            + (long)mbase * CIN_ + c0;
        unsigned long long gaddr = (unsigned long long)(uintptr_t)src;   // bytes
        unsigned lds = (unsigned)(unsigned long long)(uintptr_t)&Alds[buf][0];
        u32x4 g0;
        g0[0] = 1u;                                             // count=1, user mode
        g0[1] = lds;                                            // lds_addr
        g0[2] = (unsigned)gaddr;                                // global_addr[31:0]
        g0[3] = (unsigned)((gaddr >> 32) & 0x01FFFFFFull)       // global_addr[56:32]
              | (2u << 30);                                     // type = 2 (image)
        u32x8 g1;
        g1[0] = 1u << 16;            // data_size = 2 bytes, no multicast/pad/iter
        g1[1] = 512u << 16;          // tensor_dim0[15:0]  = 512 (cin row length)
        g1[2] = 128u << 16;          // tensor_dim1[15:0]  = 128
        g1[3] = 32u  << 16;          // tile_dim0 = 32
        g1[4] = 128u;                // tile_dim1 = 128, tile_dim2 = 0
        g1[5] = 512u;                // tensor_dim0_stride = 512 elements
        g1[6] = 0u;
        g1[7] = 0u;
        asm volatile("tensor_load_to_lds %0, %1" :: "s"(g0), "s"(g1) : "memory");
    };

    // --- B tile: 32 cins x 128 pixels from padded bf16 x, LDS [pixel][k] ---
    auto stageB = [&](int chunk, int buf) {
        int tap = chunk >> 4;
        int c0  = (chunk & 15) << 5;
        int dh  = tap / 3;           // 0..2 (already includes +1 pad offset)
        int dw  = tap - dh * 3;
        unsigned short* bl = &Blds[buf][0];
        #pragma unroll
        for (int it = 0; it < 8; ++it) {
            int pairIdx = tid + it * 256;       // 0..2047
            int c2 = pairIdx >> 7;              // cin pair 0..15
            int p  = pairIdx & 127;             // pixel in tile
            int c  = c2 << 1;
            long base = (((long)(b * CIN_ + c0 + c)) * HP_
                         + (h0 + (p >> 6) + dh)) * WP_ + (p & 63) + dw;
            unsigned u0 = xpad[base];
            unsigned u1 = xpad[base + (long)PLANE_];
            *(unsigned*)&bl[p * 32 + c] = (u1 << 16) | u0;
        }
    };

    auto compute = [&](int buf) {
        union Frag { u32x4 u[2]; v16bf v; };
        Frag afrag[2], bfrag[4];
        const unsigned short* al = &Alds[buf][0];
        const unsigned short* bl = &Blds[buf][0];
        #pragma unroll
        for (int mi = 0; mi < 2; ++mi) {
            int row = wm * 32 + mi * 16 + lrow;
            afrag[mi].u[0] = *(const u32x4*)&al[row * 32 + lhi * 8];
            afrag[mi].u[1] = *(const u32x4*)&al[row * 32 + 16 + lhi * 8];
        }
        #pragma unroll
        for (int nj = 0; nj < 4; ++nj) {
            int pix = wn * 64 + nj * 16 + lrow;
            bfrag[nj].u[0] = *(const u32x4*)&bl[pix * 32 + lhi * 8];
            bfrag[nj].u[1] = *(const u32x4*)&bl[pix * 32 + 16 + lhi * 8];
        }
        #pragma unroll
        for (int mi = 0; mi < 2; ++mi)
            #pragma unroll
            for (int nj = 0; nj < 4; ++nj)
                acc[mi][nj] = __builtin_amdgcn_wmma_f32_16x16x32_bf16(
                    false, afrag[mi].v, false, bfrag[nj].v,
                    (short)0, acc[mi][nj], false, false);
    };

    // ---- software pipeline: double-buffered LDS, one barrier per chunk ----
    if (wave == 0) tdmA(0, 0);
    stageB(0, 0);
    for (int chunk = 0; chunk < NCHUNK_; ++chunk) {
        int cur = chunk & 1;
        if (wave == 0) __builtin_amdgcn_s_wait_tensorcnt(0);  // A(chunk) landed
        __syncthreads();                                      // B visible; buf^1 free
        if (chunk + 1 < NCHUNK_) {
            if (wave == 0) tdmA(chunk + 1, cur ^ 1);
            stageB(chunk + 1, cur ^ 1);
        }
        compute(cur);
    }

    // ---- epilogue: C/D layout -> M = lhi*8 + v, N = lrow ----
    const int h = h0 + wn;
    #pragma unroll
    for (int mi = 0; mi < 2; ++mi) {
        int cout0 = mbase + wm * 32 + mi * 16 + lhi * 8;
        #pragma unroll
        for (int nj = 0; nj < 4; ++nj) {
            int w = nj * 16 + lrow;
            long base = (((long)b * COUT_ + cout0) * HW_ + h) * HW_ + w;
            #pragma unroll
            for (int v = 0; v < 8; ++v)
                out[base + (long)v * HW_ * HW_] = acc[mi][nj][v] + bias[cout0 + v];
        }
    }
}

// ---------------------------------------------------------------------------
// launch
// ---------------------------------------------------------------------------
extern "C" void kernel_launch(void* const* d_in, const int* in_sizes, int n_in,
                              void* d_out, int out_size, void* d_ws, size_t ws_size,
                              hipStream_t stream) {
    const float* x      = (const float*)d_in[0];
    const float* style  = (const float*)d_in[1];
    const float* weight = (const float*)d_in[2];
    const float* bias   = (const float*)d_in[3];
    const float* mod_w  = (const float*)d_in[4];
    const float* mod_b  = (const float*)d_in[5];
    float* out = (float*)d_out;

    char* ws = (char*)d_ws;
    float* s     = (float*)ws;                         // 32 KB
    float* demod = (float*)(ws + 32 * 1024);           // 32 KB
    unsigned short* wprep = (unsigned short*)(ws + 64 * 1024);          // 75.5 MB
    size_t wprep_bytes = (size_t)B_ * 9 * COUT_ * CIN_ * sizeof(unsigned short);
    unsigned short* xpad = (unsigned short*)(ws + 64 * 1024 + wprep_bytes); // 77.9 MB

    style_mod_kernel<<<(B_ * CIN_) / 256, 256, 0, stream>>>(style, mod_w, mod_b, s);
    demod_kernel<<<(B_ * COUT_) / 8, 256, 0, stream>>>(weight, s, demod);
    prep_weights_kernel<<<8192, 256, 0, stream>>>(weight, s, demod, wprep);
    pad_x_kernel<<<B_ * CIN_, 256, 0, stream>>>(x, xpad);

    dim3 grid(HW_ * HW_ / 128, COUT_ / 128, B_);   // (32, 4, 16)
    modconv_wmma_kernel<<<grid, 256, 0, stream>>>(xpad, wprep, bias, out);
}
